// LocalCrossAttentionBlock3D_72859825209379
// MI455X (gfx1250) — compile-verified, matrix operand order checked by reading, not run on hardware
//
#include <hip/hip_runtime.h>
#include <hip/hip_bf16.h>

typedef __attribute__((ext_vector_type(16))) _Float16 v16h;
typedef __attribute__((ext_vector_type(8)))  float    v8f;

constexpr int CH    = 96;        // channels
constexpr int HID   = 384;       // FFN hidden
constexpr int DHW   = 110592;    // 48^3
constexpr int NWINB = 13824;     // windows per batch (24^3)
constexpr int NWIN  = 27648;     // total windows (B=2)

// Packed-weight layout offsets (in _Float16 units, per stream).
// Each B-fragment = 32 lanes x 16 halves, contiguous:  [tile][lane][j]
constexpr int PK_Wi     = 0;                 // 18 tiles x 3 kc
constexpr int PK_Wo     = PK_Wi + 18 * 3 * 512;   // 27648
constexpr int PK_W1     = PK_Wo + 6 * 3 * 512;    // 36864
constexpr int PK_W2     = PK_W1 + 24 * 3 * 512;   // 73728
constexpr int PK_CW     = PK_W2 + 6 * 12 * 512;   // 110592
constexpr int PK_STRIDE = PK_CW + 6 * 3 * 512;    // 119808 per stream

// A/B fragment element map for v_wmma_f32_16x16x32_f16 (16-bit, K=32):
// lane group g = lane>>4 ; packed half j in [0,16) -> K index
__device__ __forceinline__ int kmap(int j, int g) {
  int p = j >> 1, h = j & 1;
  int base = (p < 4) ? (p * 2) : (16 + (p - 4) * 2);
  return base + h + g * 8;
}

__device__ __forceinline__ float gelu_erf(float x) {
  return 0.5f * x * (1.0f + erff(x * 0.70710678118654752f));
}

__device__ __forceinline__ v16h load_bfrag(const _Float16* base, int tk, int lane) {
  return *(const v16h*)(base + ((long)(tk * 32 + lane)) * 16);
}

struct AParams {
  const float* x[2];
  const float* qn_w[2];  const float* qn_b[2];
  const float* kvn_w[2]; const float* kvn_b[2];
  const float* bi[2];
  const float* bo[2];
  const float* fn_w[2];  const float* fn_b[2];
  const float* b1[2];
  const float* b2[2];
  const _Float16* wpk;
  float* xtok;
};

// ---------------------------------------------------------------------------
// Kernel W: pre-pack all weight matrices (f32 row-major) into f16 WMMA
// B-fragment order. Runs once per launch; output is L2-resident thereafter.
// ---------------------------------------------------------------------------
__global__ __launch_bounds__(256) void pack_weights_kernel(
    const float* Wi0, const float* Wo0, const float* W10, const float* W20, const float* cw0,
    const float* Wi1, const float* Wo1, const float* W11, const float* W21, const float* cw1,
    _Float16* __restrict__ wpk) {
  int idx = blockIdx.x * 256 + threadIdx.x;
  if (idx >= 2 * PK_STRIDE) return;
  int s = idx / PK_STRIDE;
  int r = idx - s * PK_STRIDE;
  const float* Wi = s ? Wi1 : Wi0;
  const float* Wo = s ? Wo1 : Wo0;
  const float* W1 = s ? W11 : W10;
  const float* W2 = s ? W21 : W20;
  const float* cw = s ? cw1 : cw0;

  int region, base;
  if      (r < PK_Wo) { region = 0; base = PK_Wi; }
  else if (r < PK_W1) { region = 1; base = PK_Wo; }
  else if (r < PK_W2) { region = 2; base = PK_W1; }
  else if (r < PK_CW) { region = 3; base = PK_W2; }
  else                { region = 4; base = PK_CW; }
  int rr   = r - base;
  int j    = rr & 15;
  int lane = (rr >> 4) & 31;
  int tk   = rr >> 9;
  int g = lane >> 4, mm = lane & 15;
  int kidx = kmap(j, g);

  float v;
  if (region == 0)      { int t = tk / 3,  kc  = tk % 3;  v = Wi[(t * 16 + mm) * CH  + kc  * 32 + kidx]; }
  else if (region == 1) { int t = tk / 3,  kc  = tk % 3;  v = Wo[(t * 16 + mm) * CH  + kc  * 32 + kidx]; }
  else if (region == 2) { int t = tk / 3,  kc  = tk % 3;  v = W1[(t * 16 + mm) * CH  + kc  * 32 + kidx]; }
  else if (region == 3) { int nt = tk / 12, kc2 = tk % 12; v = W2[(nt * 16 + mm) * HID + kc2 * 32 + kidx]; }
  else                  { int t = tk / 3,  kc  = tk % 3;  v = cw[(t * 16 + mm) * CH  + kc  * 32 + kidx]; }
  wpk[idx] = (_Float16)v;
}

// ---------------------------------------------------------------------------
// Kernel A: fused window transformer block (both streams, cross-attention).
// Block = 64 threads (2 waves). 4 window-pairs per block; each wave owns a
// 16-row (2-window) WMMA tile for both fg and bg.
// ---------------------------------------------------------------------------
__global__ __launch_bounds__(64) void attn_ffn_kernel(AParams P) {
  __shared__ float    s_x[2][32][CH];        // residual streams (f32)
  __shared__ _Float16 s_qkv[32][3 * CH];     // packed Q|K|V for current stream
  __shared__ float    s_scr[2][16][CH];      // per-wave scratch (attnO / hidden)
  __shared__ float    s_p[2][2][4][8][8];    // per-wave softmax probs

  const int tid  = threadIdx.x;
  const int lane = tid & 31;
  const int wv   = tid >> 5;      // wave id (0..1)
  const int g    = lane >> 4;     // lane half-group
  const int m    = lane & 15;     // tile row (A/D frag) / tile col (B/D frag)
  const int row  = wv * 16 + m;   // LDS row for this lane's A-fragment row
  const int win0 = blockIdx.x * 4;

  // ---- global -> LDS: gather 4 windows x 8 tokens x 96ch for both streams
  for (int e = tid; e < 2 * 32 * CH; e += 64) {
    int st = e / (32 * CH);
    int r2 = e - st * 32 * CH;
    int rr = r2 / CH;
    int c  = r2 - rr * CH;
    int wg = win0 + (rr >> 3);
    int l  = rr & 7;
    int w2 = wg % 24, h2 = (wg / 24) % 24, d2 = (wg / 576) % 24, b = wg / NWINB;
    int d = d2 * 2 + (l >> 2), h = h2 * 2 + ((l >> 1) & 1), ww = w2 * 2 + (l & 1);
    long off = (long)(b * CH + c) * DHW + (d * 48 + h) * 48 + ww;
    s_x[st][rr][c] = P.x[st][off];
  }
  __syncthreads();

  // ---- pre-residual kv-LayerNorm A-fragments for BOTH streams (registers)
  v16h kvA[2][3];
#pragma unroll
  for (int s = 0; s < 2; ++s) {
    float s1 = 0.f, s2 = 0.f;
    for (int c = g * 48; c < g * 48 + 48; ++c) {
      float v = s_x[s][row][c]; s1 += v; s2 += v * v;
    }
    s1 += __shfl_xor(s1, 16); s2 += __shfl_xor(s2, 16);
    float mean = s1 * (1.f / 96.f);
    float rstd = rsqrtf(s2 * (1.f / 96.f) - mean * mean + 1e-5f);
#pragma unroll
    for (int kc = 0; kc < 3; ++kc)
#pragma unroll
      for (int j = 0; j < 16; ++j) {
        int k = kc * 32 + kmap(j, g);
        float v = (s_x[s][row][k] - mean) * rstd * P.kvn_w[s][k] + P.kvn_b[s][k];
        kvA[s][kc][j] = (_Float16)v;
      }
  }

  const float scale = 0.20412414523193154f;  // 1/sqrt(24)

  // ==================== attention (per stream, fully specialized) ==========
#pragma unroll
  for (int s = 0; s < 2; ++s) {
    const int o = 1 - s;
    const _Float16* wq = P.wpk + s * PK_STRIDE + PK_Wi;
    const _Float16* wo = P.wpk + s * PK_STRIDE + PK_Wo;

    // q-LayerNorm A-fragments
    v16h qA[3];
    {
      float s1 = 0.f, s2 = 0.f;
      for (int c = g * 48; c < g * 48 + 48; ++c) {
        float v = s_x[s][row][c]; s1 += v; s2 += v * v;
      }
      s1 += __shfl_xor(s1, 16); s2 += __shfl_xor(s2, 16);
      float mean = s1 * (1.f / 96.f);
      float rstd = rsqrtf(s2 * (1.f / 96.f) - mean * mean + 1e-5f);
#pragma unroll
      for (int kc = 0; kc < 3; ++kc)
#pragma unroll
        for (int j = 0; j < 16; ++j) {
          int k = kc * 32 + kmap(j, g);
          qA[kc][j] = (_Float16)((s_x[s][row][k] - mean) * rstd * P.qn_w[s][k] + P.qn_b[s][k]);
        }
    }

    // Q projection tiles (nt 0..5): D = X_qln @ Wq^T + bq
    for (int nt = 0; nt < 6; ++nt) {
      v8f acc;
      const float bia = P.bi[s][nt * 16 + m];
#pragma unroll
      for (int r = 0; r < 8; ++r) acc[r] = bia;
#pragma unroll
      for (int kc = 0; kc < 3; ++kc) {
        v16h B = load_bfrag(wq, nt * 3 + kc, lane);
        acc = __builtin_amdgcn_wmma_f32_16x16x32_f16(false, qA[kc], false, B,
                                                     (short)0, acc, false, false);
      }
#pragma unroll
      for (int r = 0; r < 8; ++r)
        s_qkv[wv * 16 + r + 8 * g][nt * 16 + m] = (_Float16)acc[r];
    }

    // K/V projection tiles (nt 6..17): input is OTHER stream's kv-LN frags
    for (int nt = 6; nt < 18; ++nt) {
      v8f acc;
      const float bia = P.bi[s][nt * 16 + m];
#pragma unroll
      for (int r = 0; r < 8; ++r) acc[r] = bia;
#pragma unroll
      for (int kc = 0; kc < 3; ++kc) {
        v16h B = load_bfrag(wq, nt * 3 + kc, lane);
        acc = __builtin_amdgcn_wmma_f32_16x16x32_f16(false, kvA[o][kc], false, B,
                                                     (short)0, acc, false, false);
      }
#pragma unroll
      for (int r = 0; r < 8; ++r)
        s_qkv[wv * 16 + r + 8 * g][nt * 16 + m] = (_Float16)acc[r];
    }

    // scores + softmax (8x8 per head, dh=24): lane = (i<<2)|jgroup
    {
      const int iL = lane >> 2, jg = lane & 3;
      for (int wl = 0; wl < 2; ++wl) {
        const int rb = wv * 16 + wl * 8;
        for (int h = 0; h < 4; ++h) {
          float sc0 = 0.f, sc1 = 0.f;
          for (int d = 0; d < 24; ++d) {
            float qv = (float)s_qkv[rb + iL][h * 24 + d];
            sc0 += qv * (float)s_qkv[rb + jg * 2][CH + h * 24 + d];
            sc1 += qv * (float)s_qkv[rb + jg * 2 + 1][CH + h * 24 + d];
          }
          sc0 *= scale; sc1 *= scale;
          float mx = fmaxf(sc0, sc1);
          mx = fmaxf(mx, __shfl_xor(mx, 1));
          mx = fmaxf(mx, __shfl_xor(mx, 2));
          float e0 = __expf(sc0 - mx), e1 = __expf(sc1 - mx);
          float sm = e0 + e1;
          sm += __shfl_xor(sm, 1);
          sm += __shfl_xor(sm, 2);
          float inv = 1.f / sm;
          s_p[wv][wl][h][iL][jg * 2]     = e0 * inv;
          s_p[wv][wl][h][iL][jg * 2 + 1] = e1 * inv;
        }
      }
    }

    // O = P @ V  -> per-wave scratch
    for (int idx = lane; idx < 2 * 8 * CH; idx += 32) {
      int wl = idx / (8 * CH);
      int r2 = idx - wl * 8 * CH;
      int i = r2 / CH, c = r2 - (r2 / CH) * CH;
      int h = c / 24;
      int rb = wv * 16 + wl * 8;
      float a = 0.f;
      for (int j = 0; j < 8; ++j)
        a += s_p[wv][wl][h][i][j] * (float)s_qkv[rb + j][2 * CH + c];
      s_scr[wv][wl * 8 + i][c] = a;
    }

    // out-projection + residual into s_x[s]
    v16h oA[3];
#pragma unroll
    for (int kc = 0; kc < 3; ++kc)
#pragma unroll
      for (int j = 0; j < 16; ++j)
        oA[kc][j] = (_Float16)s_scr[wv][m][kc * 32 + kmap(j, g)];
    for (int nt = 0; nt < 6; ++nt) {
      v8f acc;
      const float bov = P.bo[s][nt * 16 + m];
#pragma unroll
      for (int r = 0; r < 8; ++r) acc[r] = bov;
#pragma unroll
      for (int kc = 0; kc < 3; ++kc) {
        v16h B = load_bfrag(wo, nt * 3 + kc, lane);
        acc = __builtin_amdgcn_wmma_f32_16x16x32_f16(false, oA[kc], false, B,
                                                     (short)0, acc, false, false);
      }
#pragma unroll
      for (int r = 0; r < 8; ++r)
        s_x[s][wv * 16 + r + 8 * g][nt * 16 + m] += acc[r];
    }
  }

  // ==================== FFN (per stream, fully specialized) ================
#pragma unroll
  for (int s = 0; s < 2; ++s) {
    const _Float16* w1 = P.wpk + s * PK_STRIDE + PK_W1;
    const _Float16* w2 = P.wpk + s * PK_STRIDE + PK_W2;
    v16h hA[3];
    {
      float s1 = 0.f, s2 = 0.f;
      for (int c = g * 48; c < g * 48 + 48; ++c) {
        float v = s_x[s][row][c]; s1 += v; s2 += v * v;
      }
      s1 += __shfl_xor(s1, 16); s2 += __shfl_xor(s2, 16);
      float mean = s1 * (1.f / 96.f);
      float rstd = rsqrtf(s2 * (1.f / 96.f) - mean * mean + 1e-5f);
#pragma unroll
      for (int kc = 0; kc < 3; ++kc)
#pragma unroll
        for (int j = 0; j < 16; ++j) {
          int k = kc * 32 + kmap(j, g);
          hA[kc][j] = (_Float16)((s_x[s][row][k] - mean) * rstd * P.fn_w[s][k] + P.fn_b[s][k]);
        }
    }
    v8f accf[6];
#pragma unroll
    for (int nt = 0; nt < 6; ++nt) {
      float bb = P.b2[s][nt * 16 + m];
#pragma unroll
      for (int r = 0; r < 8; ++r) accf[nt][r] = bb;
    }
    // stream hidden in 32-wide K-chunks: FFN1 (+GELU) feeds FFN2 immediately
    for (int kc2 = 0; kc2 < 12; ++kc2) {
#pragma unroll
      for (int sub = 0; sub < 2; ++sub) {
        const int th = kc2 * 2 + sub;              // W1 tile (row block of 384)
        v8f hacc;
        const float b1v = P.b1[s][th * 16 + m];
#pragma unroll
        for (int r = 0; r < 8; ++r) hacc[r] = b1v;
#pragma unroll
        for (int kc = 0; kc < 3; ++kc) {
          v16h B = load_bfrag(w1, th * 3 + kc, lane);
          hacc = __builtin_amdgcn_wmma_f32_16x16x32_f16(false, hA[kc], false, B,
                                                        (short)0, hacc, false, false);
        }
#pragma unroll
        for (int r = 0; r < 8; ++r)
          s_scr[wv][r + 8 * g][sub * 16 + m] = gelu_erf(hacc[r]);
      }
      // repack the 16x32 hidden chunk as an A fragment, accumulate FFN2
      v16h aH;
#pragma unroll
      for (int j = 0; j < 16; ++j)
        aH[j] = (_Float16)s_scr[wv][m][kmap(j, g)];
#pragma unroll
      for (int nt = 0; nt < 6; ++nt) {
        v16h B = load_bfrag(w2, nt * 12 + kc2, lane);
        accf[nt] = __builtin_amdgcn_wmma_f32_16x16x32_f16(false, aH, false, B,
                                                          (short)0, accf[nt], false, false);
      }
    }
#pragma unroll
    for (int nt = 0; nt < 6; ++nt)
#pragma unroll
      for (int r = 0; r < 8; ++r)
        s_x[s][wv * 16 + r + 8 * g][nt * 16 + m] += accf[nt][r];
  }

  // ---- LDS -> token-major workspace [slab=st*2+b][token][96]
  __syncthreads();
  for (int e = tid; e < 2 * 32 * CH; e += 64) {
    int st = e / (32 * CH);
    int r2 = e - st * 32 * CH;
    int rr = r2 / CH;
    int c  = r2 - rr * CH;
    int wg = win0 + (rr >> 3);
    int l  = rr & 7;
    int b  = wg / NWINB;
    int wb = wg - b * NWINB;
    long t = (long)wb * 8 + l;
    P.xtok[(((long)(st * 2 + b)) * DHW + t) * CH + c] = s_x[st][rr][c];
  }
}

// ---------------------------------------------------------------------------
// Kernel B: 1x1x1 conv as WMMA GEMM + per-(b,c) sum/sumsq reduction.
// Writes un-normalized conv output into d_out (window-reversed layout).
// ---------------------------------------------------------------------------
__global__ __launch_bounds__(128) void conv_stats_kernel(
    const float* __restrict__ xtok, const _Float16* __restrict__ wpk,
    const float* cb0, const float* cb1,
    float* __restrict__ out, float* __restrict__ gsum, float* __restrict__ gsq) {
  __shared__ float sstat[96][2];
  const int tid = threadIdx.x, lane = tid & 31, wv = tid >> 5;
  const int g = lane >> 4, m = lane & 15;
  const int slab = blockIdx.y;            // st*2 + b
  const int st = slab >> 1, b = slab & 1;
  const long base_t = (long)blockIdx.x * 64 + wv * 16;
  const _Float16* cwp = wpk + st * PK_STRIDE + PK_CW;
  const float* cb = st ? cb1 : cb0;

  for (int i = tid; i < 192; i += 128) (&sstat[0][0])[i] = 0.f;
  __syncthreads();

  v16h A[3];
  const float* xrow = xtok + ((long)slab * DHW + base_t + m) * CH;
#pragma unroll
  for (int kc = 0; kc < 3; ++kc)
#pragma unroll
    for (int j = 0; j < 16; ++j)
      A[kc][j] = (_Float16)xrow[kc * 32 + kmap(j, g)];

  for (int nt = 0; nt < 6; ++nt) {
    const int n = nt * 16 + m;
    v8f acc;
    const float bia = cb[n];
#pragma unroll
    for (int r = 0; r < 8; ++r) acc[r] = bia;
#pragma unroll
    for (int kc = 0; kc < 3; ++kc) {
      v16h B = load_bfrag(cwp, nt * 3 + kc, lane);
      acc = __builtin_amdgcn_wmma_f32_16x16x32_f16(false, A[kc], false, B,
                                                   (short)0, acc, false, false);
    }
    float ps = 0.f, pq = 0.f;
#pragma unroll
    for (int r = 0; r < 8; ++r) {
      float y = acc[r];
      long t = base_t + r + 8 * g;
      int win = (int)(t >> 3), l = (int)(t & 7);
      int w2 = win % 24, h2 = (win / 24) % 24, d2 = win / 576;
      int d = d2 * 2 + (l >> 2), h = h2 * 2 + ((l >> 1) & 1), ww = w2 * 2 + (l & 1);
      long off = (long)st * (2L * CH * DHW) + (long)(b * CH + n) * DHW +
                 (d * 48 + h) * 48 + ww;
      out[off] = y;
      ps += y; pq += y * y;
    }
    ps += __shfl_xor(ps, 16);
    pq += __shfl_xor(pq, 16);
    if (g == 0) {
      atomicAdd(&sstat[n][0], ps);
      atomicAdd(&sstat[n][1], pq);
    }
  }
  __syncthreads();
  for (int c = tid; c < 96; c += 128) {
    atomicAdd(&gsum[slab * 96 + c], sstat[c][0]);
    atomicAdd(&gsq[slab * 96 + c],  sstat[c][1]);
  }
}

// ---------------------------------------------------------------------------
// Kernel C: InstanceNorm (affine) + erf-GELU, in place on d_out.
// ---------------------------------------------------------------------------
__global__ __launch_bounds__(256) void norm_gelu_kernel(
    float* __restrict__ out, const float* __restrict__ gsum,
    const float* __restrict__ gsq,
    const float* iw0, const float* ib0, const float* iw1, const float* ib1) {
  long idx = (long)blockIdx.x * 256 + threadIdx.x;
  const long TOT = 4L * CH * DHW;
  if (idx >= TOT) return;
  const long per_st = 2L * CH * DHW;
  int st = (int)(idx / per_st);
  long r = idx - (long)st * per_st;
  int b = (int)(r / ((long)CH * DHW));
  long r2 = r - (long)b * CH * DHW;
  int c = (int)(r2 / DHW);
  int slab = st * 2 + b;
  float mean = gsum[slab * 96 + c] * (1.f / DHW);
  float var  = gsq[slab * 96 + c] * (1.f / DHW) - mean * mean;
  float rstd = rsqrtf(var + 1e-5f);
  const float* iw = st ? iw1 : iw0;
  const float* ib = st ? ib1 : ib0;
  float y = out[idx];
  y = (y - mean) * rstd * iw[c] + ib[c];
  out[idx] = gelu_erf(y);
}

extern "C" void kernel_launch(void* const* d_in, const int* in_sizes, int n_in,
                              void* d_out, int out_size, void* d_ws, size_t ws_size,
                              hipStream_t stream) {
  (void)in_sizes; (void)n_in; (void)out_size; (void)ws_size;
  auto IN = [&](int s, int k) -> const float* {
    return (const float*)d_in[2 + s * 18 + k];
  };

  // workspace layout
  float* xtok = (float*)d_ws;
  const size_t XT = 4UL * DHW * CH * sizeof(float);            // 169,869,312 B
  _Float16* wpk = (_Float16*)((char*)d_ws + XT);               // 479,232 B
  const size_t WPK = 2UL * PK_STRIDE * sizeof(_Float16);
  float* gsum = (float*)((char*)d_ws + XT + WPK);
  float* gsq  = gsum + 4 * 96;

  AParams P;
  P.x[0] = (const float*)d_in[0];
  P.x[1] = (const float*)d_in[1];
  for (int s = 0; s < 2; ++s) {
    P.qn_w[s]  = IN(s, 0);  P.qn_b[s]  = IN(s, 1);
    P.kvn_w[s] = IN(s, 2);  P.kvn_b[s] = IN(s, 3);
    P.bi[s]    = IN(s, 5);
    P.bo[s]    = IN(s, 7);
    P.fn_w[s]  = IN(s, 8);  P.fn_b[s]  = IN(s, 9);
    P.b1[s]    = IN(s, 11);
    P.b2[s]    = IN(s, 13);
  }
  P.wpk  = wpk;
  P.xtok = xtok;

  hipMemsetAsync(gsum, 0, 2 * 4 * 96 * sizeof(float), stream);

  pack_weights_kernel<<<dim3((2 * PK_STRIDE + 255) / 256), dim3(256), 0, stream>>>(
      IN(0, 4), IN(0, 6), IN(0, 10), IN(0, 12), IN(0, 14),
      IN(1, 4), IN(1, 6), IN(1, 10), IN(1, 12), IN(1, 14), wpk);

  attn_ffn_kernel<<<dim3(NWIN / 4), dim3(64), 0, stream>>>(P);

  conv_stats_kernel<<<dim3(DHW / 64, 4), dim3(128), 0, stream>>>(
      xtok, wpk, IN(0, 15), IN(1, 15), (float*)d_out, gsum, gsq);

  const long tot = 4L * CH * DHW;
  norm_gelu_kernel<<<dim3((unsigned)((tot + 255) / 256)), dim3(256), 0, stream>>>(
      (float*)d_out, gsum, gsq, IN(0, 16), IN(0, 17), IN(1, 16), IN(1, 17));
}
